// Decoder_35364760715949
// MI455X (gfx1250) — compile-verified, matrix operand order checked by reading, not run on hardware
//
#include <hip/hip_runtime.h>
#include <stdint.h>

// ---------------- problem constants ----------------
#define BATCH 32
#define T_IN  256
#define T_OUT 400
#define EDIM  512
#define ADIM  128
#define NFLT  32
#define KSZ   31
#define PRE   256
#define DDIM  1024
#define NMEL  80
#define K0    1792          // 768 (l0_k) + 1024 (l0_r)
#define K1    2560          // 1536 (l1_k) + 1024 (l1_r)
#define N4D   4096
#define GRID  64            // persistent-kernel blocks (kept small so all are resident)
#define BLK   256           // 8 waves of 32
#define NWAVE 8

#define MEL_BASE   0
#define GATE_BASE  ((size_t)BATCH * T_OUT * NMEL)                 // 1,024,000
#define ALIGN_BASE (GATE_BASE + (size_t)BATCH * T_OUT)            // 1,036,800

// ---------------- WMMA types ----------------
typedef __attribute__((ext_vector_type(16))) __bf16 v16bf;
typedef __attribute__((ext_vector_type(8)))  float  v8f;
union Frag { uint4 q[2]; v16bf v; };

__device__ __forceinline__ uint16_t f2bf(float x) {
  union { float f; uint32_t u; } v; v.f = x;
  uint32_t r = v.u + 0x7FFFu + ((v.u >> 16) & 1u);   // round-to-nearest-even
  return (uint16_t)(r >> 16);
}
__device__ __forceinline__ float sigm(float x) { return 1.f / (1.f + __expf(-x)); }

// ---------------- grid-wide barrier (persistent kernel) ----------------
__device__ __forceinline__ void gridbar(unsigned* bar) {
  __syncthreads();
  __threadfence();                                    // release prior global writes
  if (threadIdx.x == 0) {
    unsigned g = __hip_atomic_load(&bar[1], __ATOMIC_RELAXED, __HIP_MEMORY_SCOPE_AGENT);
    unsigned a = __hip_atomic_fetch_add(&bar[0], 1u, __ATOMIC_ACQ_REL, __HIP_MEMORY_SCOPE_AGENT);
    if (a == GRID - 1) {
      __hip_atomic_store(&bar[0], 0u,   __ATOMIC_RELAXED, __HIP_MEMORY_SCOPE_AGENT);
      __hip_atomic_store(&bar[1], g+1u, __ATOMIC_RELEASE, __HIP_MEMORY_SCOPE_AGENT);
    } else {
      while (__hip_atomic_load(&bar[1], __ATOMIC_ACQUIRE, __HIP_MEMORY_SCOPE_AGENT) == g)
        __builtin_amdgcn_s_sleep(2);
    }
  }
  __syncthreads();
  __threadfence();                                    // acquire for all lanes
}

// ---------------- pre-pass kernels ----------------
// Swizzle fp32 weights (concat [Wk; Wr], K x 4096) into WMMA B-fragment order, bf16.
// Fragment layout per 16x16x32 tile: lane L holds column n*16+(L&15), elements j:
//   j<8 -> K = kb*32 + 8*(L>>4) + j ; j>=8 -> K = kb*32 + 16 + 8*(L>>4) + (j-8)
__global__ void k_swizzle_w(const float* __restrict__ Wk, const float* __restrict__ Wr,
                            int krows, uint16_t* __restrict__ out, int nelem) {
  int gid = blockIdx.x * BLK + threadIdx.x;
  if (gid >= nelem) return;
  int j    = gid & 15;
  int lane = (gid >> 4) & 31;
  int tile = gid >> 9;
  int nt = tile & 255, kb = tile >> 8;
  int lh = lane >> 4;
  int kk = (j < 8) ? (j + lh * 8) : (16 + (j - 8) + lh * 8);
  int k  = kb * 32 + kk;
  int col = nt * 16 + (lane & 15);
  float v = (k < krows) ? Wk[(size_t)k * N4D + col]
                        : Wr[(size_t)(k - krows) * N4D + col];
  out[gid] = f2bf(v);
}

// Prenet for all (b, t): one block per decoder frame; output bf16.
__global__ void k_prenet(const float* __restrict__ targets,
                         const float* __restrict__ fc1w, const float* __restrict__ fc1b,
                         const float* __restrict__ fc2w, const float* __restrict__ fc2b,
                         uint16_t* __restrict__ pre_out) {
  __shared__ float din[NMEL];
  __shared__ float hbuf[PRE];
  int bt = blockIdx.x;                 // 0 .. B*T_OUT-1
  int b = bt / T_OUT, t = bt % T_OUT;
  int tid = threadIdx.x;               // 0..255
  if (tid < NMEL)
    din[tid] = (t == 0) ? 0.f : targets[((size_t)b * T_OUT + (t - 1)) * NMEL + tid];
  __syncthreads();
  float h = fc1b[tid];
  for (int k = 0; k < NMEL; ++k) h += din[k] * fc1w[k * PRE + tid];
  hbuf[tid] = fmaxf(h, 0.f);
  __syncthreads();
  float o = fc2b[tid];
  for (int k = 0; k < PRE; ++k) o += hbuf[k] * fc2w[k * PRE + tid];
  pre_out[((size_t)b * T_OUT + t) * PRE + tid] = f2bf(fmaxf(o, 0.f));
}

// keys_proc = memory @ m_w + m_b   -> [B*T_IN, A] fp32
__global__ void k_keys(const float* __restrict__ mem, const float* __restrict__ mw,
                       const float* __restrict__ mb, float* __restrict__ keys) {
  int gid = blockIdx.x * BLK + threadIdx.x;
  if (gid >= BATCH * T_IN * ADIM) return;
  int a = gid & (ADIM - 1);
  int bt = gid >> 7;
  float acc = mb[a];
  const float* mrow = mem + (size_t)bt * EDIM;
  for (int e = 0; e < EDIM; ++e) acc += mrow[e] * mw[e * ADIM + a];
  keys[gid] = acc;
}

// ---------------- persistent scan kernel ----------------
struct ScanParams {
  const float* memory; const uint8_t* mask;
  const float* l0_b; const float* l1_b;
  const float* q_w;  const float* q_b;
  const float* v_w;  const float* v_b;
  const float* lcw;  const float* lcb;      // loc conv
  const float* ldw;  const float* ldb;      // loc dense
  const float* proj_w; const float* proj_b;
  const float* gate_w; const float* gate_b;
  uint16_t *A0, *A1, *W0, *W1, *pre;        // bf16 buffers
  float *c0, *c1, *wcum, *z0, *z1, *h1f, *ctxf, *qp, *ebuf, *keys;
  unsigned* bar;
  float* out;
};

// One WMMA output tile per wave: 512 waves == 2 Mtiles x 256 Ntiles exactly.
// A (32 x K bf16, row-major) staged through an 8 KB LDS panel per 128 K-columns,
// copied with CDNA5 async global->LDS DMA (bypasses VGPRs, tracked on ASYNCcnt).
__device__ __forceinline__ void gemm_tiles(const uint16_t* __restrict__ Amat, int K,
                                           const uint16_t* __restrict__ W,
                                           float* __restrict__ Z,
                                           uint16_t* s_panel) {
  const int tid  = threadIdx.x;
  const int wave = tid >> 5, lane = tid & 31;
  const int gw   = blockIdx.x * NWAVE + wave;     // 0..511
  const int mt   = gw & 1, nt = gw >> 1;          // Mtile, Ntile
  const int lrow = lane & 15, lh = lane >> 4;
  v8f acc = {0.f, 0.f, 0.f, 0.f, 0.f, 0.f, 0.f, 0.f};
  const int npan = K >> 7;                        // panels of 128 K-cols
  const int r = tid >> 3, co = (tid & 7) << 4;    // panel staging split
  for (int p = 0; p < npan; ++p) {
    __syncthreads();
    { // async-stage 32x128 bf16 A panel (8 KB): 32 B per thread, no VGPR bounce.
      // ISA 15.18: INST_OFFSET is added to BOTH the LDS and global addresses,
      // so the two issues below move bytes [0,16) and [16,32) of this slice.
      uint32_t lds_off = (uint32_t)(uintptr_t)(s_panel + r * 128 + co);
      const uint16_t* g = Amat + (size_t)r * K + p * 128 + co;
      asm volatile("global_load_async_to_lds_b128 %0, %1, off"
                   :: "v"(lds_off), "v"(g) : "memory");
      asm volatile("global_load_async_to_lds_b128 %0, %1, off offset:16"
                   :: "v"(lds_off), "v"(g) : "memory");
      asm volatile("s_wait_asynccnt 0x0" ::: "memory");
    }
    __syncthreads();
#pragma unroll
    for (int kb = 0; kb < 4; ++kb) {
      Frag af, bf;
      // A fragment: row = mt*16+lrow; elems 0-7 = K kb*32+8*lh.., elems 8-15 = +16
      const uint4* ap = (const uint4*)(s_panel + (mt * 16 + lrow) * 128 + kb * 32 + lh * 8);
      af.q[0] = ap[0]; af.q[1] = ap[2];
      // B fragment: pre-swizzled, wave reads 1 KB contiguous
      const size_t tileIdx = (size_t)(p * 4 + kb) * 256 + nt;
      const uint4* bp = (const uint4*)(W + (tileIdx * 32 + lane) * 16);
      bf.q[0] = bp[0]; bf.q[1] = bp[1];
      acc = __builtin_amdgcn_wmma_f32_16x16x32_bf16(false, af.v, false, bf.v,
                                                    (short)0, acc, false, false);
    }
    if (p + 1 < npan) {  // prefetch next weight panel (global_prefetch_b8)
      const size_t nTile = (size_t)((p + 1) * 4) * 256 + nt;
      __builtin_prefetch(W + (nTile * 32 + lane) * 16, 0, 1);
    }
  }
#pragma unroll
  for (int rr = 0; rr < 8; ++rr) {   // C layout: VGPR rr -> M = rr + 8*lh
    int row = mt * 16 + lh * 8 + rr;
    int col = nt * 16 + lrow;
    Z[(size_t)row * N4D + col] = acc[rr];
  }
}

__global__ __launch_bounds__(BLK) void k_scan(ScanParams P) {
  __shared__ uint16_t s_panel[32 * 128];   // 8 KB GEMM A panel
  __shared__ float s_red[T_IN];
  __shared__ float s_w[T_IN];
  const int tid  = threadIdx.x;
  const int gtid = blockIdx.x * BLK + tid;

  // fill prenet slot of A0 for t = 0 (ctx/h slots are zeroed by memset)
  if (gtid < BATCH * PRE) {
    int b = gtid >> 8, j = gtid & 255;
    P.A0[b * K0 + j] = P.pre[((size_t)b * T_OUT) * PRE + j];
  }
  gridbar(P.bar);

  for (int t = 0; t < T_OUT; ++t) {
    // ---- LSTM0 GEMM: z0 = [x_t, ctx, h0] @ [l0_k; l0_r] ----
    gemm_tiles(P.A0, K0, P.W0, P.z0, s_panel);
    gridbar(P.bar);
    // ---- LSTM0 activation ----
    for (int i = gtid; i < BATCH * DDIM; i += GRID * BLK) {
      int b = i >> 10, d = i & 1023;
      const float* z = P.z0 + (size_t)b * N4D;
      float zi = z[d]            + P.l0_b[d];
      float zf = z[DDIM + d]     + P.l0_b[DDIM + d];
      float zg = z[2 * DDIM + d] + P.l0_b[2 * DDIM + d];
      float zo = z[3 * DDIM + d] + P.l0_b[3 * DDIM + d];
      float c = sigm(zf) * P.c0[i] + sigm(zi) * tanhf(zg);
      float h = sigm(zo) * tanhf(c);
      P.c0[i] = c;
      uint16_t hb = f2bf(h);
      P.A0[b * K0 + 768 + d] = hb;   // next step's LSTM0 input
      P.A1[b * K1 + d]       = hb;   // this step's LSTM1 input
    }
    gridbar(P.bar);
    // ---- LSTM1 GEMM: z1 = [h0, ctx, h1] @ [l1_k; l1_r] ----
    gemm_tiles(P.A1, K1, P.W1, P.z1, s_panel);
    gridbar(P.bar);
    // ---- LSTM1 activation ----
    for (int i = gtid; i < BATCH * DDIM; i += GRID * BLK) {
      int b = i >> 10, d = i & 1023;
      const float* z = P.z1 + (size_t)b * N4D;
      float zi = z[d]            + P.l1_b[d];
      float zf = z[DDIM + d]     + P.l1_b[DDIM + d];
      float zg = z[2 * DDIM + d] + P.l1_b[2 * DDIM + d];
      float zo = z[3 * DDIM + d] + P.l1_b[3 * DDIM + d];
      float c = sigm(zf) * P.c1[i] + sigm(zi) * tanhf(zg);
      float h = sigm(zo) * tanhf(c);
      P.c1[i] = c;
      P.h1f[i] = h;                          // fp32 query / proj input
      P.A1[b * K1 + 1536 + d] = f2bf(h);     // next step's LSTM1 input
    }
    gridbar(P.bar);
    // ---- query projection qp = h1 @ q_w + q_b ----
    if (gtid < BATCH * ADIM) {
      int b = gtid >> 7, a = gtid & 127;
      float acc = P.q_b[a];
      const float* h1 = P.h1f + (size_t)b * DDIM;
      for (int d = 0; d < DDIM; ++d) acc += h1[d] * P.q_w[d * ADIM + a];
      P.qp[gtid] = acc;
    }
    gridbar(P.bar);
    // ---- location conv + energies: one thread per (b, ti) ----
    if (gtid < BATCH * T_IN) {
      int b = gtid >> 8, ti = gtid & 255;
      float lf[NFLT];
#pragma unroll
      for (int nf = 0; nf < NFLT; ++nf) lf[nf] = P.lcb[nf];
      for (int k = 0; k < KSZ; ++k) {
        int tt = ti + k - (KSZ / 2);
        if (tt >= 0 && tt < T_IN) {
          float wv = P.wcum[b * T_IN + tt];
#pragma unroll
          for (int nf = 0; nf < NFLT; ++nf) lf[nf] += wv * P.lcw[k * NFLT + nf];
        }
      }
      float acc = P.v_b[0];
      const float* krow = P.keys + (size_t)gtid * ADIM;
      for (int a = 0; a < ADIM; ++a) {
        float s = P.qp[b * ADIM + a] + krow[a] + P.ldb[a];
#pragma unroll
        for (int nf = 0; nf < NFLT; ++nf) s += lf[nf] * P.ldw[nf * ADIM + a];
        acc += P.v_w[a] * tanhf(s);
      }
      if (!P.mask[gtid]) acc += -1e9f;
      P.ebuf[gtid] = acc;
    }
    gridbar(P.bar);
    // ---- softmax + context: block b handles batch row b ----
    if (blockIdx.x < BATCH) {
      int b = blockIdx.x, ti = tid;
      float ev = P.ebuf[b * T_IN + ti];
      s_red[ti] = ev; __syncthreads();
      for (int o = T_IN / 2; o > 0; o >>= 1) {
        if (ti < o) s_red[ti] = fmaxf(s_red[ti], s_red[ti + o]);
        __syncthreads();
      }
      float mx = s_red[0]; __syncthreads();
      float pr = __expf(ev - mx);
      s_red[ti] = pr; __syncthreads();
      for (int o = T_IN / 2; o > 0; o >>= 1) {
        if (ti < o) s_red[ti] += s_red[ti + o];
        __syncthreads();
      }
      float w = pr / s_red[0];
      s_w[ti] = w;
      P.wcum[b * T_IN + ti] += w;
      P.out[ALIGN_BASE + ((size_t)b * T_OUT + t) * T_IN + ti] = w;
      __syncthreads();
      for (int eo = ti; eo < EDIM; eo += BLK) {
        float acc = 0.f;
        const float* mcol = P.memory + ((size_t)b * T_IN) * EDIM + eo;
        for (int tt = 0; tt < T_IN; ++tt) acc += s_w[tt] * mcol[(size_t)tt * EDIM];
        P.ctxf[b * EDIM + eo] = acc;
        uint16_t cb = f2bf(acc);
        P.A0[b * K0 + PRE + eo]  = cb;     // next step's LSTM0 input
        P.A1[b * K1 + DDIM + eo] = cb;     // next step's LSTM1 input
      }
    }
    gridbar(P.bar);
    // ---- outputs (mel, gate) + next-step prenet fill ----
    if (gtid < BATCH * NMEL) {
      int b = gtid / NMEL, m = gtid % NMEL;
      float acc = P.proj_b[m];
      const float* h1 = P.h1f  + (size_t)b * DDIM;
      const float* cx = P.ctxf + (size_t)b * EDIM;
      for (int k = 0; k < DDIM; ++k) acc += h1[k] * P.proj_w[k * NMEL + m];
      for (int k = 0; k < EDIM; ++k) acc += cx[k] * P.proj_w[(DDIM + k) * NMEL + m];
      P.out[((size_t)b * T_OUT + t) * NMEL + m] = acc;
    } else if (gtid < BATCH * NMEL + BATCH) {
      int b = gtid - BATCH * NMEL;
      float acc = P.gate_b[0];
      const float* h1 = P.h1f  + (size_t)b * DDIM;
      const float* cx = P.ctxf + (size_t)b * EDIM;
      for (int k = 0; k < DDIM; ++k) acc += h1[k] * P.gate_w[k];
      for (int k = 0; k < EDIM; ++k) acc += cx[k] * P.gate_w[DDIM + k];
      P.out[GATE_BASE + (size_t)b * T_OUT + t] = acc;
    } else if (gtid >= 4096 && gtid < 4096 + BATCH * PRE && t + 1 < T_OUT) {
      int g2 = gtid - 4096;
      int b = g2 >> 8, j = g2 & 255;
      P.A0[b * K0 + j] = P.pre[((size_t)b * T_OUT + (t + 1)) * PRE + j];
    }
    gridbar(P.bar);
  }
}

// ---------------- launcher ----------------
extern "C" void kernel_launch(void* const* d_in, const int* in_sizes, int n_in,
                              void* d_out, int out_size, void* d_ws, size_t ws_size,
                              hipStream_t stream) {
  const float*   memory  = (const float*)d_in[0];
  const float*   targets = (const float*)d_in[1];
  const uint8_t* mask    = (const uint8_t*)d_in[2];
  const float* p_fc1_w = (const float*)d_in[3];
  const float* p_fc1_b = (const float*)d_in[4];
  const float* p_fc2_w = (const float*)d_in[5];
  const float* p_fc2_b = (const float*)d_in[6];
  const float* l0_k = (const float*)d_in[7];
  const float* l0_r = (const float*)d_in[8];
  const float* l0_b = (const float*)d_in[9];
  const float* l1_k = (const float*)d_in[10];
  const float* l1_r = (const float*)d_in[11];
  const float* l1_b = (const float*)d_in[12];
  const float* q_w  = (const float*)d_in[13];
  const float* q_b  = (const float*)d_in[14];
  const float* m_w  = (const float*)d_in[15];
  const float* m_b  = (const float*)d_in[16];
  const float* v_w  = (const float*)d_in[17];
  const float* v_b  = (const float*)d_in[18];
  const float* lcw  = (const float*)d_in[19];
  const float* lcb  = (const float*)d_in[20];
  const float* ldw  = (const float*)d_in[21];
  const float* ldb  = (const float*)d_in[22];
  const float* proj_w = (const float*)d_in[23];
  const float* proj_b = (const float*)d_in[24];
  const float* gate_w = (const float*)d_in[25];
  const float* gate_b = (const float*)d_in[26];

  char* ws = (char*)d_ws;
  size_t off = 0;
  auto alloc = [&](size_t bytes) -> void* {
    void* p = ws + off;
    off += (bytes + 255) & ~(size_t)255;
    return p;
  };
  // --- zero-initialized region first (one async memset) ---
  uint16_t* A0  = (uint16_t*)alloc((size_t)BATCH * K0 * 2);
  uint16_t* A1  = (uint16_t*)alloc((size_t)BATCH * K1 * 2);
  float* c0     = (float*)alloc((size_t)BATCH * DDIM * 4);
  float* c1     = (float*)alloc((size_t)BATCH * DDIM * 4);
  float* wcum   = (float*)alloc((size_t)BATCH * T_IN * 4);
  unsigned* bar = (unsigned*)alloc(256);
  size_t zeroBytes = off;
  // --- rest of workspace ---
  uint16_t* W0  = (uint16_t*)alloc((size_t)K0 * N4D * 2);    // 14.7 MB
  uint16_t* W1  = (uint16_t*)alloc((size_t)K1 * N4D * 2);    // 21.0 MB
  uint16_t* pre = (uint16_t*)alloc((size_t)BATCH * T_OUT * PRE * 2);
  float* keys   = (float*)alloc((size_t)BATCH * T_IN * ADIM * 4);
  float* z0     = (float*)alloc((size_t)BATCH * N4D * 4);
  float* z1     = (float*)alloc((size_t)BATCH * N4D * 4);
  float* h1f    = (float*)alloc((size_t)BATCH * DDIM * 4);
  float* ctxf   = (float*)alloc((size_t)BATCH * EDIM * 4);
  float* qp     = (float*)alloc((size_t)BATCH * ADIM * 4);
  float* ebuf   = (float*)alloc((size_t)BATCH * T_IN * 4);
  (void)in_sizes; (void)n_in; (void)out_size; (void)ws_size;

  hipMemsetAsync(d_ws, 0, zeroBytes, stream);

  const int n0 = (K0 / 32) * 256 * 512;   // 7,340,032 fragment elements
  k_swizzle_w<<<(n0 + BLK - 1) / BLK, BLK, 0, stream>>>(l0_k, l0_r, 768, W0, n0);
  const int n1 = (K1 / 32) * 256 * 512;   // 10,485,760
  k_swizzle_w<<<(n1 + BLK - 1) / BLK, BLK, 0, stream>>>(l1_k, l1_r, 1536, W1, n1);
  k_prenet<<<BATCH * T_OUT, PRE, 0, stream>>>(targets, p_fc1_w, p_fc1_b,
                                              p_fc2_w, p_fc2_b, pre);
  k_keys<<<(BATCH * T_IN * ADIM + BLK - 1) / BLK, BLK, 0, stream>>>(memory, m_w, m_b, keys);

  ScanParams P;
  P.memory = memory; P.mask = mask;
  P.l0_b = l0_b; P.l1_b = l1_b;
  P.q_w = q_w; P.q_b = q_b; P.v_w = v_w; P.v_b = v_b;
  P.lcw = lcw; P.lcb = lcb; P.ldw = ldw; P.ldb = ldb;
  P.proj_w = proj_w; P.proj_b = proj_b; P.gate_w = gate_w; P.gate_b = gate_b;
  P.A0 = A0; P.A1 = A1; P.W0 = W0; P.W1 = W1; P.pre = pre;
  P.c0 = c0; P.c1 = c1; P.wcum = wcum; P.z0 = z0; P.z1 = z1;
  P.h1f = h1f; P.ctxf = ctxf; P.qp = qp; P.ebuf = ebuf; P.keys = keys;
  P.bar = bar; P.out = (float*)d_out;

  k_scan<<<GRID, BLK, 0, stream>>>(P);
}
